// DecoderMHA_2525440770892
// MI455X (gfx1250) — matrix operand order repaired and checked by reading, not compiled, not run on hardware
//
#include <hip/hip_runtime.h>
#include <hip/hip_bf16.h>

typedef __attribute__((ext_vector_type(16))) __bf16 v16bf;
typedef __attribute__((ext_vector_type(8)))  float  v8f;

#define CC 1024           // cities
#define EE 128            // embed
#define HH 8              // heads (head_dim = 16 == one WMMA N-tile)
#define NW 16             // waves per block
#define NT 512            // threads per block (16 wave32)
#define MTW 4             // M-tiles (of 16 rows) per wave: 16*4*16 = 1024 rows
#define WSTRIDE 144       // padded LDS row stride (bf16 elems) for bank spread
#define INF_ 1.0e8f

// Load one 16x128 bf16 A-strip (rows c0..c0+15 of h[b]) into 4 WMMA A fragments,
// converting f32->bf16 in registers. A 16x32 k order used here (B reads the same order):
// lane = row (l%16), half (l/16) selects contiguous K {16h..16h+15} per K-step.
__device__ __forceinline__ void load_a_strip(const float* __restrict__ rowp, v16bf afr[4]) {
#pragma unroll
  for (int t = 0; t < 8; ++t) {
    const float4 x0 = *(const float4*)(rowp + 8 * t + 16 * (t >> 1));
    const float4 x1 = *(const float4*)(rowp + 8 * t + 16 * (t >> 1) + 4);
    const int f = t >> 1;
    const int o = (t & 1) * 8;
    afr[f][o + 0] = (__bf16)x0.x; afr[f][o + 1] = (__bf16)x0.y;
    afr[f][o + 2] = (__bf16)x0.z; afr[f][o + 3] = (__bf16)x0.w;
    afr[f][o + 4] = (__bf16)x1.x; afr[f][o + 5] = (__bf16)x1.y;
    afr[f][o + 6] = (__bf16)x1.z; afr[f][o + 7] = (__bf16)x1.w;
  }
}

__global__ __launch_bounds__(NT)
void decoder_mha_kernel(const float* __restrict__ h,     const float* __restrict__ hN,
                        const float* __restrict__ hPrev, const float* __restrict__ h0,
                        const int*   __restrict__ mask,
                        const float* __restrict__ Qw,  const float* __restrict__ Qb,
                        const float* __restrict__ Vw,  const float* __restrict__ Vb,
                        const float* __restrict__ Wow, const float* __restrict__ Wob,
                        const float* __restrict__ Qfw, const float* __restrict__ Qfb,
                        const float* __restrict__ Kfw, const float* __restrict__ Kfb,
                        float* __restrict__ out) {
  __shared__ alignas(32) __bf16 s_w[EE * WSTRIDE];   // Vw bf16 (B-layout friendly rows)
  __shared__ alignas(32) __bf16 s_u[16 * WSTRIDE];   // U.T: per-head q-folded V weights
  __shared__ float s_minf[CC];                        // mask * 1e8
  __shared__ float s_hc[3 * EE];
  __shared__ float s_q[EE];                           // q * (1/sqrt(16))
  __shared__ float s_qf[EE];                          // qf * (1/sqrt(128))
  __shared__ float s_ctx[EE];
  __shared__ float s_dec[EE];
  __shared__ float s_bu[16];                          // q_h . Vb_h per head (0 for 8..15)
  __shared__ float s_ukf[EE];                         // Kfw.T @ qf
  __shared__ float s_bkf;                             // qf . Kfb
  __shared__ float s_wS[NW][32];                      // per-wave per-lane exp-sum partials
  __shared__ float s_wctx[NW][HH][16];                // per-wave context accumulators

  const int b    = blockIdx.x;
  const int tid  = threadIdx.x;
  const int w    = tid >> 5;
  const int lane = tid & 31;
  const int l16  = lane & 15;
  const int half = lane >> 4;
  const float* hB = h + (size_t)b * CC * EE;

  // ---- Stage: h_c, mask, Vw(bf16); zero U and context accumulators ----
  if (tid < 3 * EE) {
    const float* src = (tid < EE)     ? (hN    + (size_t)b * EE + tid)
                     : (tid < 2 * EE) ? (hPrev + (size_t)b * EE + tid - EE)
                                      : (h0    + (size_t)b * EE + tid - 2 * EE);
    s_hc[tid] = *src;
  }
  for (int c = tid; c < CC; c += NT) s_minf[c] = (float)mask[(size_t)b * CC + c] * INF_;
  for (int i = tid; i < EE * EE; i += NT) s_w[(i >> 7) * WSTRIDE + (i & 127)] = (__bf16)Vw[i];
  for (int i = tid; i < 16 * WSTRIDE; i += NT) s_u[i] = (__bf16)0.f;
  for (int i = tid; i < NW * HH * 16; i += NT) ((float*)s_wctx)[i] = 0.f;
  __syncthreads();

  // ---- q = h_c @ Qw.T + Qb, pre-scaled by 1/sqrt(head_dim) ----
  if (tid < EE) {
    float acc = Qb[tid];
    const float* qr = Qw + (size_t)tid * 3 * EE;
#pragma unroll 4
    for (int k = 0; k < 3 * EE; ++k) acc += s_hc[k] * qr[k];
    s_q[tid] = acc * 0.25f;
  }
  __syncthreads();

  // ---- Fold q into V weights: U[hd][k] = sum_j q[16hd+j]*Vw[16hd+j][k] ----
  for (int idx = tid; idx < EE * HH; idx += NT) {
    const int k = idx & 127, hd = idx >> 7;
    float a = 0.f;
#pragma unroll
    for (int j = 0; j < 16; ++j) a += s_q[hd * 16 + j] * Vw[(size_t)(hd * 16 + j) * EE + k];
    s_u[hd * WSTRIDE + k] = (__bf16)a;
  }
  if (tid < HH) {
    float a = 0.f;
#pragma unroll
    for (int j = 0; j < 16; ++j) a += s_q[tid * 16 + j] * Vb[tid * 16 + j];
    s_bu[tid] = a;
  }
  if (tid >= HH && tid < 16) s_bu[tid] = 0.f;
  __syncthreads();

  // ---- Pass 1: per M-tile: scores for all heads via one GEMM, then weighted
  //      accumulation of v = h@Vw.T (bias folded analytically at combine) ----
  float sacc = 0.f;
#pragma unroll 1
  for (int mt = 0; mt < MTW; ++mt) {
    const int c0 = w * (MTW * 16) + mt * 16;
    v16bf afr[4];
    load_a_strip(hB + (size_t)(c0 + l16) * EE + 16 * half, afr);

    float msk[8];
#pragma unroll
    for (int p = 0; p < 8; ++p) msk[p] = s_minf[c0 + p + 8 * half];

    // scores: dacc[p] = dot(head=l16) for row p+8*half
    v8f dacc = {};
#pragma unroll
    for (int ks = 0; ks < 4; ++ks) {
      const v16bf bfr = *(const v16bf*)&s_u[l16 * WSTRIDE + ks * 32 + 16 * half];
      dacc = __builtin_amdgcn_wmma_f32_16x16x32_bf16(false, afr[ks], false, bfr,
                                                     (short)0, dacc, false, false);
    }
    const float bu = s_bu[l16];
    float e[8];
#pragma unroll
    for (int p = 0; p < 8; ++p) {
      e[p] = __expf(dacc[p] + bu - msk[p]);   // fixed-max softmax numerator
      sacc += e[p];
    }

#pragma unroll 1
    for (int nt = 0; nt < HH; ++nt) {         // one N-tile == one head
      v8f acc = {};
#pragma unroll
      for (int ks = 0; ks < 4; ++ks) {
        const v16bf bfr = *(const v16bf*)&s_w[(nt * 16 + l16) * WSTRIDE + ks * 32 + 16 * half];
        acc = __builtin_amdgcn_wmma_f32_16x16x32_bf16(false, afr[ks], false, bfr,
                                                      (short)0, acc, false, false);
      }
      const int src = nt + (lane & 16);       // lane holding this head's weights (per half)
      float cacc = 0.f;
#pragma unroll
      for (int p = 0; p < 8; ++p) cacc += __shfl(e[p], src, 32) * acc[p];
      cacc += __shfl_xor(cacc, 16, 32);       // combine row halves
      if (half == 0) s_wctx[w][nt][l16] += cacc;
    }
  }
  s_wS[w][lane] = sacc;
  __syncthreads();

  // ---- Combine: context = sum(e*v)/sum(e) + Vb (bias folded) ----
  if (tid < EE) {
    const int hd = tid >> 4, n = tid & 15;
    float S = 0.f, cx = 0.f;
#pragma unroll
    for (int ww = 0; ww < NW; ++ww) {
      S  += s_wS[ww][hd] + s_wS[ww][hd + 16];
      cx += s_wctx[ww][hd][n];
    }
    s_ctx[tid] = cx / S + Vb[tid];
  }
  __syncthreads();

  // ---- dec_out = context @ Wow.T + Wob ; qf = dec_out @ Qfw.T + Qfb ----
  if (tid < EE) {
    float acc = Wob[tid];
    const float* wr = Wow + (size_t)tid * EE;
#pragma unroll 4
    for (int k = 0; k < EE; ++k) acc += s_ctx[k] * wr[k];
    s_dec[tid] = acc;
  }
  __syncthreads();
  if (tid < EE) {
    float acc = Qfb[tid];
    const float* wr = Qfw + (size_t)tid * EE;
#pragma unroll 4
    for (int k = 0; k < EE; ++k) acc += s_dec[k] * wr[k];
    s_qf[tid] = acc * 0.0883883476483184f;    // 1/sqrt(128)
  }
  __syncthreads();

  // ---- Fold qf into Kf weights: ukf = Kfw.T @ qf ; bkf = qf . Kfb ----
  if (tid < EE) {
    float a = 0.f;
#pragma unroll 4
    for (int e = 0; e < EE; ++e) a += s_qf[e] * Kfw[(size_t)e * EE + tid];
    s_ukf[tid] = a;
  }
  if (tid == 0) {
    float bk = 0.f;
#pragma unroll 4
    for (int e = 0; e < EE; ++e) bk += s_qf[e] * Kfb[e];
    s_bkf = bk;
  }
  __syncthreads();

  // ---- Pass 2: logits[c] = h[c].ukf + bkf ; out = 10*tanh - mask*INF ----
#pragma unroll 1
  for (int c = tid; c < CC; c += NT) {
    const float* rp = hB + (size_t)c * EE;
    float dot = 0.f;
#pragma unroll 8
    for (int k = 0; k < EE; k += 4) {
      const float4 x = *(const float4*)(rp + k);
      dot += x.x * s_ukf[k] + x.y * s_ukf[k + 1] + x.z * s_ukf[k + 2] + x.w * s_ukf[k + 3];
    }
    out[(size_t)b * CC + c] = 10.f * tanhf(dot + s_bkf) - s_minf[c];
  }
}

extern "C" void kernel_launch(void* const* d_in, const int* in_sizes, int n_in,
                              void* d_out, int out_size, void* d_ws, size_t ws_size,
                              hipStream_t stream) {
  const float* h     = (const float*)d_in[0];
  const float* hN    = (const float*)d_in[1];
  const float* hPrev = (const float*)d_in[2];
  const float* h0    = (const float*)d_in[3];
  const int*   mask  = (const int*)d_in[4];
  const float* Qw    = (const float*)d_in[5];
  const float* Qb    = (const float*)d_in[6];
  const float* Vw    = (const float*)d_in[7];
  const float* Vb    = (const float*)d_in[8];
  const float* Wow   = (const float*)d_in[9];
  const float* Wob   = (const float*)d_in[10];
  const float* Qfw   = (const float*)d_in[11];
  const float* Qfb   = (const float*)d_in[12];
  const float* Kfw   = (const float*)d_in[13];
  const float* Kfb   = (const float*)d_in[14];
  float* out = (float*)d_out;

  const int Bn = in_sizes[0] / (CC * EE);
  if (Bn <= 0) return;
  hipLaunchKernelGGL(decoder_mha_kernel, dim3(Bn), dim3(NT), 0, stream,
                     h, hN, hPrev, h0, mask, Qw, Qb, Vw, Vb, Wow, Wob,
                     Qfw, Qfb, Kfw, Kfb, out);
}
